// CausalSelfAttention_62285615726781
// MI455X (gfx1250) — matrix operand order repaired and checked by reading, not compiled, not run on hardware
//
#include <hip/hip_runtime.h>
#include <hip/hip_bf16.h>

typedef __attribute__((ext_vector_type(16))) _Float16 v16h;
typedef __attribute__((ext_vector_type(8)))  _Float16 v8h;
typedef __attribute__((ext_vector_type(4)))  _Float16 v4h;
typedef __attribute__((ext_vector_type(8)))  float    v8f;

// Problem constants (B=4, T=2048, C=1024, H=16, D=64)
#define TB 4
#define TT 2048
#define TC 1024
#define TH 16
#define TD 64
#define GM (TB*TT)   // 8192
#define GN TC        // 1024
#define GK TC        // 1024

__device__ __forceinline__ v8f wmma_f16(v16h a, v16h b, v8f c) {
    return __builtin_amdgcn_wmma_f32_16x16x32_f16(false, a, false, b,
                                                  (short)0, c, false, false);
}

__device__ __forceinline__ v16h cat16(v8h a, v8h b) {
    return __builtin_shufflevector(a, b, 0,1,2,3,4,5,6,7,8,9,10,11,12,13,14,15);
}

// ---------------------------------------------------------------------------
// fp32 -> f16 bulk convert, 4 elements per thread (b128 in, b64 out)
// ---------------------------------------------------------------------------
__global__ __launch_bounds__(256) void cvt_kernel(
    const float* __restrict__ in, _Float16* __restrict__ out)
{
    const int i = blockIdx.x * blockDim.x + threadIdx.x;
    const float4 v = ((const float4*)in)[i];
    v4h o;
    o[0] = (_Float16)v.x; o[1] = (_Float16)v.y;
    o[2] = (_Float16)v.z; o[3] = (_Float16)v.w;
    ((v4h*)out)[i] = o;
}

// ---------------------------------------------------------------------------
// GEMM: out = X(GM x GK, f16) * W^T(W is GN x GK row-major, f16) + bias
// mode 0: fp32 out row-major [GM, GN]     (final projection -> d_out)
// mode 1: f16 out as [B,H,T,D]            (q, k)
// mode 2: f16 out as [B,H,D,T]            (transposed V)
// One wave per 32x32 tile: per K-step (32) 8x b128 loads feed 4 WMMAs,
// doubling flop/byte vs a 16x16 tile.
// ---------------------------------------------------------------------------
__global__ __launch_bounds__(256) void gemm_kernel(
    const _Float16* __restrict__ X, const _Float16* __restrict__ W,
    const float* __restrict__ bias, float* __restrict__ outf,
    _Float16* __restrict__ outh, int mode)
{
    const int wave = threadIdx.x >> 5;
    const int lane = threadIdx.x & 31;
    const int tile = blockIdx.x * 8 + wave;      // 0 .. 8191
    const int tn = tile & (GN/32 - 1);           // 0 .. 31
    const int tm = tile >> 5;                    // 0 .. 255
    const int mloc = lane & 15;
    const int hi   = lane >> 4;

    const _Float16* arow0 = X + (size_t)(tm*32 + mloc) * GK;       // rows 0..15
    const _Float16* arow1 = X + (size_t)(tm*32 + 16 + mloc) * GK;  // rows 16..31
    const _Float16* brow0 = W + (size_t)(tn*32 + mloc) * GK;       // cols 0..15
    const _Float16* brow1 = W + (size_t)(tn*32 + 16 + mloc) * GK;  // cols 16..31

    v8f acc00 = {}, acc01 = {}, acc10 = {}, acc11 = {};
    for (int k = 0; k < GK; k += 32) {
        // A 16x32 f16 layout: halves 0..7 -> K=hi*8+j ; 8..15 -> K=16+hi*8+j
        const v16h a0 = cat16(*(const v8h*)(arow0 + k + hi*8),
                              *(const v8h*)(arow0 + k + 16 + hi*8));
        const v16h a1 = cat16(*(const v8h*)(arow1 + k + hi*8),
                              *(const v8h*)(arow1 + k + 16 + hi*8));
        // B 32x16 f16 layout: halves 0..7 -> K=hi*8+j ; 8..15 -> K=16+hi*8+j
        const v16h b0 = cat16(*(const v8h*)(brow0 + k + hi*8),
                              *(const v8h*)(brow0 + k + 16 + hi*8));
        const v16h b1 = cat16(*(const v8h*)(brow1 + k + hi*8),
                              *(const v8h*)(brow1 + k + 16 + hi*8));
        acc00 = wmma_f16(a0, b0, acc00);
        acc01 = wmma_f16(a0, b1, acc01);
        acc10 = wmma_f16(a1, b0, acc10);
        acc11 = wmma_f16(a1, b1, acc11);
    }

    #pragma unroll
    for (int i = 0; i < 2; ++i) {
        #pragma unroll
        for (int j = 0; j < 2; ++j) {
            const v8f acc = (i == 0) ? (j == 0 ? acc00 : acc01)
                                     : (j == 0 ? acc10 : acc11);
            const int n = tn*32 + j*16 + mloc;
            const float bn = bias[n];
            #pragma unroll
            for (int r = 0; r < 8; ++r) {
                const int m = tm*32 + i*16 + r + 8*hi;  // C layout: M = r+8*hi
                const float val = acc[r] + bn;
                if (mode == 0) {
                    outf[(size_t)m * GN + n] = val;
                } else {
                    const int b = m >> 11, t = m & (TT - 1);
                    const int h = n >> 6,  d = n & (TD - 1);
                    size_t idx;
                    if (mode == 1) idx = (((size_t)(b*TH + h)*TT) + t)*TD + d;
                    else           idx = (((size_t)(b*TH + h)*TD) + d)*TT + t;
                    outh[idx] = (_Float16)val;
                }
            }
        }
    }
}

// ---------------------------------------------------------------------------
// RoPE in-place on f16 [B,H,T,D]: one thread per (row, pair), fp32 math
// ---------------------------------------------------------------------------
__global__ __launch_bounds__(256) void rope_kernel(_Float16* __restrict__ q)
{
    const int idx = blockIdx.x * blockDim.x + threadIdx.x;  // B*H*T*32 threads
    const int i   = idx & 31;
    const int row = idx >> 5;
    const int t   = row & (TT - 1);
    const float freq = __expf(-0.2878231366f * (float)i);   // 10000^(-2i/64)
    const float ang  = (float)t * freq;
    const float c = __cosf(ang), s = __sinf(ang);
    _Float16* p = q + (size_t)row * TD + 2*i;
    const float x0 = (float)p[0], x1 = (float)p[1];
    p[0] = (_Float16)(x0*c - x1*s);
    p[1] = (_Float16)(x0*s + x1*c);
}

// ---------------------------------------------------------------------------
// Flash attention: one wave per 16-query tile of one (b,h). All f16 in/out.
// Q: [B,H,T,D], K: [B,H,T,D], VT: [B,H,D,T], Y out: f16 [B,T,C]
// ---------------------------------------------------------------------------
__global__ __launch_bounds__(256) void attn_kernel(
    const _Float16* __restrict__ Q, const _Float16* __restrict__ K,
    const _Float16* __restrict__ VT, _Float16* __restrict__ Y)
{
    __shared__ __align__(16) _Float16 lds_p[8][16*32];  // per-wave P staging

    const int wave = threadIdx.x >> 5;
    const int lane = threadIdx.x & 31;
    const int tile = blockIdx.x * 8 + wave;     // 0 .. 8191
    const int qt = tile & (TT/16 - 1);          // query tile 0..127
    const int bh = tile >> 7;                   // b*H + h
    const int t0 = qt * 16;
    const int mloc = lane & 15;
    const int hi   = lane >> 4;

    // Q tile (16x64) as two A fragments: pure b128 loads
    const _Float16* qrow = Q + ((size_t)bh * TT + t0 + mloc) * TD;
    const v16h aq0 = cat16(*(const v8h*)(qrow + hi*8),
                           *(const v8h*)(qrow + 16 + hi*8));
    const v16h aq1 = cat16(*(const v8h*)(qrow + 32 + hi*8),
                           *(const v8h*)(qrow + 48 + hi*8));

    v8f acc0 = {}, acc1 = {}, acc2 = {}, acc3 = {};
    float rmax[8], rsum[8];
    #pragma unroll
    for (int r = 0; r < 8; ++r) { rmax[r] = -3.0e38f; rsum[r] = 0.0f; }

    const float scale = 0.125f;                 // 1/sqrt(64), applied to scores
    const int kbmax = (t0 + 15) >> 5;
    for (int kb = 0; kb <= kbmax; ++kb) {
        const int s0 = kb * 32;

        // ---- scores S = Q*K^T for 32 keys: two 16x16 fp32 tiles ----
        v8f sc0, sc1;
        {
            const _Float16* kr0 = K + ((size_t)bh*TT + s0 + mloc) * TD;
            const _Float16* kr1 = K + ((size_t)bh*TT + s0 + 16 + mloc) * TD;
            const v16h b0 = cat16(*(const v8h*)(kr0 + hi*8),
                                  *(const v8h*)(kr0 + 16 + hi*8));
            const v16h b1 = cat16(*(const v8h*)(kr0 + 32 + hi*8),
                                  *(const v8h*)(kr0 + 48 + hi*8));
            const v16h b2 = cat16(*(const v8h*)(kr1 + hi*8),
                                  *(const v8h*)(kr1 + 16 + hi*8));
            const v16h b3 = cat16(*(const v8h*)(kr1 + 32 + hi*8),
                                  *(const v8h*)(kr1 + 48 + hi*8));
            v8f c = {};
            c   = wmma_f16(aq0, b0, c);
            sc0 = wmma_f16(aq1, b1, c);
            v8f d = {};
            d   = wmma_f16(aq0, b2, d);
            sc1 = wmma_f16(aq1, b3, d);
        }

        // ---- scale + causal mask + online softmax (rows = r + 8*hi) ----
        #pragma unroll
        for (int r = 0; r < 8; ++r) {
            const int t = t0 + r + 8*hi;
            float s0v = sc0[r] * scale;
            float s1v = sc1[r] * scale;
            if (s0 + mloc      > t) s0v = -1.0e30f;
            if (s0 + 16 + mloc > t) s1v = -1.0e30f;
            float m = fmaxf(s0v, s1v);
            #pragma unroll
            for (int x = 1; x < 16; x <<= 1)
                m = fmaxf(m, __shfl_xor(m, x, 32));
            const float mnew  = fmaxf(rmax[r], m);
            const float alpha = __expf(rmax[r] - mnew);
            const float p0 = __expf(s0v - mnew);
            const float p1 = __expf(s1v - mnew);
            sc0[r] = p0; sc1[r] = p1;
            float ps = p0 + p1;
            #pragma unroll
            for (int x = 1; x < 16; x <<= 1)
                ps += __shfl_xor(ps, x, 32);
            rsum[r] = rsum[r] * alpha + ps;
            rmax[r] = mnew;
            acc0[r] *= alpha; acc1[r] *= alpha;
            acc2[r] *= alpha; acc3[r] *= alpha;
        }

        // ---- transpose P (C layout -> A layout) via LDS, as f16 ----
        _Float16* pl = &lds_p[wave][0];
        #pragma unroll
        for (int r = 0; r < 8; ++r) {
            pl[(r + 8*hi)*32 + mloc]      = (_Float16)sc0[r];
            pl[(r + 8*hi)*32 + 16 + mloc] = (_Float16)sc1[r];
        }
        asm volatile("s_wait_dscnt 0" ::: "memory");
        __builtin_amdgcn_wave_barrier();
        const v16h ap = cat16(*(const v8h*)(pl + mloc*32 + hi*8),
                              *(const v8h*)(pl + mloc*32 + 16 + hi*8));
        __builtin_amdgcn_wave_barrier();

        // ---- O += P(16x32) * V(32x64): 4 WMMAs, V from f16 [B,H,D,T] ----
        #pragma unroll
        for (int dc = 0; dc < 4; ++dc) {
            const _Float16* vrow =
                VT + ((size_t)bh*TD + dc*16 + mloc) * TT + s0 + hi*8;
            const v16h bv = cat16(*(const v8h*)(vrow),
                                  *(const v8h*)(vrow + 16));
            if      (dc == 0) acc0 = wmma_f16(ap, bv, acc0);
            else if (dc == 1) acc1 = wmma_f16(ap, bv, acc1);
            else if (dc == 2) acc2 = wmma_f16(ap, bv, acc2);
            else              acc3 = wmma_f16(ap, bv, acc3);
        }
    }

    // ---- normalize and store y as f16 [B,T,C] with c = h*64 + d ----
    const int b = bh >> 4, h = bh & 15;
    #pragma unroll
    for (int r = 0; r < 8; ++r) {
        const float inv = 1.0f / rsum[r];
        const int t = t0 + r + 8*hi;
        const size_t base = ((size_t)(b*TT + t)) * TC + h*TD;
        Y[base +      mloc] = (_Float16)(acc0[r] * inv);
        Y[base + 16 + mloc] = (_Float16)(acc1[r] * inv);
        Y[base + 32 + mloc] = (_Float16)(acc2[r] * inv);
        Y[base + 48 + mloc] = (_Float16)(acc3[r] * inv);
    }
}

extern "C" void kernel_launch(void* const* d_in, const int* in_sizes, int n_in,
                              void* d_out, int out_size, void* d_ws, size_t ws_size,
                              hipStream_t stream)
{
    const float* x  = (const float*)d_in[0];
    const float* wq = (const float*)d_in[1];
    const float* bq = (const float*)d_in[2];
    const float* wk = (const float*)d_in[3];
    const float* bk = (const float*)d_in[4];
    const float* wv = (const float*)d_in[5];
    const float* bv = (const float*)d_in[6];
    const float* wp = (const float*)d_in[7];
    const float* bp = (const float*)d_in[8];
    float* out = (float*)d_out;

    const size_t nx = (size_t)GM * GK;          // 8,388,608
    const size_t nw = (size_t)GN * GK;          // 1,048,576
    _Float16* xh  = (_Float16*)d_ws;
    _Float16* wqh = xh  + nx;
    _Float16* wkh = wqh + nw;
    _Float16* wvh = wkh + nw;
    _Float16* wph = wvh + nw;
    _Float16* q   = wph + nw;
    _Float16* k   = q  + nx;
    _Float16* vt  = k  + nx;
    _Float16* y   = vt + nx;                    // total ~92 MB of workspace

    const int cvt_x_blocks = (int)(nx / 4 / 256);     // 8192
    const int cvt_w_blocks = (int)(nw / 4 / 256);     // 1024
    const int gemm_blocks  = (GM/32) * (GN/32) / 8;   // 1024
    const int rope_blocks  = (TB*TH*TT*(TD/2)) / 256; // 16384
    const int attn_blocks  = (TB*TH*(TT/16)) / 8;     // 1024

    cvt_kernel<<<cvt_x_blocks, 256, 0, stream>>>(x,  xh);
    cvt_kernel<<<cvt_w_blocks, 256, 0, stream>>>(wq, wqh);
    cvt_kernel<<<cvt_w_blocks, 256, 0, stream>>>(wk, wkh);
    cvt_kernel<<<cvt_w_blocks, 256, 0, stream>>>(wv, wvh);
    cvt_kernel<<<cvt_w_blocks, 256, 0, stream>>>(wp, wph);

    gemm_kernel<<<gemm_blocks, 256, 0, stream>>>(xh, wqh, bq, nullptr, q,  1);
    gemm_kernel<<<gemm_blocks, 256, 0, stream>>>(xh, wkh, bk, nullptr, k,  1);
    gemm_kernel<<<gemm_blocks, 256, 0, stream>>>(xh, wvh, bv, nullptr, vt, 2);
    rope_kernel<<<rope_blocks, 256, 0, stream>>>(q);
    rope_kernel<<<rope_blocks, 256, 0, stream>>>(k);
    attn_kernel<<<attn_blocks, 256, 0, stream>>>(q, k, vt, y);
    gemm_kernel<<<gemm_blocks, 256, 0, stream>>>(y, wph, bp, out, nullptr, 0);
}
